// SVTransformerEncoder_25555055411787
// MI455X (gfx1250) — compile-verified, matrix operand order checked by reading, not run on hardware
//
#include <hip/hip_runtime.h>
#include <hip/hip_bf16.h>
#include <math.h>

// ---------------------------------------------------------------------------
// SVTransformerEncoder on MI455X (gfx1250, wave32, WMMA).
//
// Shapes: V=128, E=256, H=4, B=1024, L=64.
// Dominant work: QKV GEMM (B*L=65536, K=256, N=3072) = 103 GFLOP -> WMMA f16.
// Fused per-(b,h) attention keeps the 800MB fp32 qkv tensor entirely in LDS
// (CDNA5: 320KB/WGP). Weights (1.5MB f16) are L2-resident (192MB L2) and are
// streamed into LDS with GLOBAL_LOAD_ASYNC_TO_LDS_B128 (ASYNCcnt-tracked,
// double-buffered) so every fragment is read from L2 exactly once per WG.
// ---------------------------------------------------------------------------

typedef __attribute__((ext_vector_type(16))) _Float16 v16h;
typedef __attribute__((ext_vector_type(8)))  _Float16 v8h;
typedef __attribute__((ext_vector_type(4)))  _Float16 v4h;
typedef __attribute__((ext_vector_type(8)))  float    v8f;

#define E_DIM 256
#define H_DIM 4
#define B_DIM 1024
#define L_DIM 64
#define F3E   768            // 3*E per head
#define LDA   264            // x2 tile LDS stride (halves), 16B aligned
#define QLD   776            // qkv tile LDS stride (halves), 16B aligned
#define SLD   68             // score tile LDS stride (floats)
#define WLD   264            // weight chunk LDS stride (halves), 16B aligned

// ---- LDS layout for the fused attention kernel (dynamic shared) -----------
#define SX_OFF     0
#define SX_BYTES   (64 * LDA * 2)                 // 33792
#define SQKV_OFF   (SX_OFF + SX_BYTES)            // 33792
#define SQKV_BYTES (64 * QLD * 2)                 // 99328
#define SS_OFF     (SQKV_OFF + SQKV_BYTES)        // 133120
#define SS_BYTES   (64 * SLD * 4)                 // 17408
#define SSC_OFF    (SS_OFF + SS_BYTES)            // 150528
#define SCC_OFF    (SSC_OFF + 64 * 4)             // 150784
#define SNW_OFF    (SCC_OFF + 64 * 4)             // 151040
#define WBUF_OFF   (SNW_OFF + 64)                 // 151104
#define WBUF_BYTES (64 * WLD * 2)                 // 33792 per buffer
#define ATTN_LDS   (WBUF_OFF + 2 * WBUF_BYTES)    // 218688 bytes < 320KB/WGP

// ---------------------------------------------------------------------------
// WMMA fragment helpers (ISA 7.12.2 layouts, 16-bit A 16x32 / B 32x16)
// ---------------------------------------------------------------------------
__device__ __forceinline__ v16h cat8(v8h lo, v8h hi) {
    return __builtin_shufflevector(lo, hi, 0,1,2,3,4,5,6,7,8,9,10,11,12,13,14,15);
}

// lane r = M (or N) index, g selects K chunk pair: K in [g*8,g*8+8) U [16+g*8,..)
__device__ __forceinline__ v16h load_frag(const _Float16* base, int stride,
                                          int row_base, int kbase, int lane) {
    const int r = lane & 15, g = lane >> 4;
    const _Float16* p = base + (size_t)(row_base + r) * stride + kbase + g * 8;
    v8h lo = *(const v8h*)p;
    v8h hi = *(const v8h*)(p + 16);
    return cat8(lo, hi);
}

__device__ __forceinline__ v8f wmma_f16(v16h a, v16h b, v8f c) {
    return __builtin_amdgcn_wmma_f32_16x16x32_f16(false, a, false, b,
                                                  (short)0, c, false, false);
}

// ---------------------------------------------------------------------------
// Async global->LDS copy (CDNA5 GLOBAL_LOAD_ASYNC_TO_LDS_B128, ASYNCcnt).
// VDST VGPR holds the per-lane LDS byte offset; generic LDS pointers carry the
// LDS offset in their low 32 bits (aperture in the high bits), so truncation
// yields the VDST value.
// ---------------------------------------------------------------------------
__device__ __forceinline__ void async_ld_b128(void* lds_ptr, const void* gptr) {
    const unsigned int lds = (unsigned int)(unsigned long long)(uintptr_t)lds_ptr;
    const unsigned long long ga = (unsigned long long)(uintptr_t)gptr;
    asm volatile("global_load_async_to_lds_b128 %0, %1, off"
                 :: "v"(lds), "v"(ga) : "memory");
}

// Stage one 64-row x 256-col f16 weight chunk into an LDS buffer (8 async
// b128 instructions per wave -> ASYNCcnt += 8 per wave).
__device__ __forceinline__ void stage_weights_async(const _Float16* Wh,
                                                    char* smemBase, int bufIdx,
                                                    int chunk, int tid) {
    _Float16* dst = (_Float16*)(smemBase + WBUF_OFF + (size_t)bufIdx * WBUF_BYTES);
    #pragma unroll
    for (int i = 0; i < 8; ++i) {
        const int c2 = i * 256 + tid;                 // 2048 16B chunks
        const int row = c2 >> 5, colh = (c2 & 31) * 8;
        async_ld_b128(dst + row * WLD + colh,
                      Wh + (size_t)(chunk * 64 + row) * E_DIM + colh);
    }
}

// ---------------------------------------------------------------------------
// Kernel 0: qkv_w f32 -> f16 (one-time; hot GEMM then does 128-bit f16 loads)
// ---------------------------------------------------------------------------
__global__ void k_convert(const float* __restrict__ w, _Float16* __restrict__ w16) {
    const int i = (blockIdx.x * 256 + threadIdx.x) * 4;
    float4 v = *(const float4*)(w + i);
    v4h h; h[0] = (_Float16)v.x; h[1] = (_Float16)v.y;
           h[2] = (_Float16)v.z; h[3] = (_Float16)v.w;
    *(v4h*)(w16 + i) = h;
}

// ---------------------------------------------------------------------------
// Kernel 1: x = LN(emb[cc]+pos); x2 = x+pos (f16); xmean = x.mean over L.
// One workgroup per batch; each wave handles 8 rows, 32 lanes cover E=256.
// ---------------------------------------------------------------------------
__global__ void k_embed_ln(const int* __restrict__ cc,
                           const float* __restrict__ emb,
                           const float* __restrict__ g0,
                           const float* __restrict__ b0,
                           _Float16* __restrict__ X2,
                           float* __restrict__ XMEAN) {
    const int b = blockIdx.x;
    const int tid = threadIdx.x, lane = tid & 31, wv = tid >> 5;
    __shared__ float sAcc[E_DIM];
    sAcc[tid] = 0.0f;
    __syncthreads();

    const float kLog = 9.210340371976184f / 128.0f;   // ln(10000)/half
    for (int i = 0; i < 8; ++i) {
        const int l = wv * 8 + i;
        const int code = cc[b * L_DIM + l];
        const float* er = emb + (size_t)code * E_DIM;
        const int e0 = lane * 8;
        float pv[8], vv[8], s = 0.0f, s2 = 0.0f;
        #pragma unroll
        for (int j = 0; j < 8; ++j) {
            const int e = e0 + j;
            const float freq = __expf(-(float)(e >> 1) * kLog);
            const float ph = (float)l * freq;
            const float p = (e & 1) ? __cosf(ph) : __sinf(ph);
            const float val = er[e] + p;
            pv[j] = p; vv[j] = val;
            s += val; s2 += val * val;
        }
        #pragma unroll
        for (int o = 16; o; o >>= 1) { s += __shfl_xor(s, o, 32); s2 += __shfl_xor(s2, o, 32); }
        const float m = s * (1.0f / 256.0f);
        const float var = s2 * (1.0f / 256.0f) - m * m;
        const float rinv = rsqrtf(var + 1e-5f);
        v8h hs;
        #pragma unroll
        for (int j = 0; j < 8; ++j) {
            const int e = e0 + j;
            const float x = (vv[j] - m) * rinv * g0[e] + b0[e];
            atomicAdd(&sAcc[e], x);                    // ds_add_f32
            hs[j] = (_Float16)(x + pv[j]);             // x2 = x + pos
        }
        *(v8h*)(X2 + ((size_t)b * L_DIM + l) * E_DIM + e0) = hs;
    }
    __syncthreads();
    XMEAN[b * E_DIM + tid] = sAcc[tid] * (1.0f / 64.0f);
}

// ---------------------------------------------------------------------------
// Kernel 2: fused QKV GEMM + attention per (b,h). 256 threads = 8 waves.
// ---------------------------------------------------------------------------
__global__ void k_attn(const int* __restrict__ cc,
                       const _Float16* __restrict__ W16,     // (3072, 256) f16
                       const _Float16* __restrict__ X2,      // (B, L, E) f16
                       float* __restrict__ AOUT) {           // (B, H, E) f32
    extern __shared__ char smem[];
    _Float16* sX    = (_Float16*)(smem + SX_OFF);
    _Float16* sQKV  = (_Float16*)(smem + SQKV_OFF);
    float*    sS    = (float*)   (smem + SS_OFF);
    float*    sScr  = (float*)   (smem + SSC_OFF);
    int*      sCC   = (int*)     (smem + SCC_OFF);
    float*    sNw   = (float*)   (smem + SNW_OFF);

    const int b = blockIdx.x >> 2, h = blockIdx.x & 3;
    const int tid = threadIdx.x, lane = tid & 31, wv = tid >> 5;
    const int r = lane & 15, g = lane >> 4;
    const _Float16* Wh = W16 + (size_t)(h * F3E) * E_DIM;

    // ---- Phase A: async-stage x2[b] tile + weight chunk 0; codes to LDS ---
    if (tid < 64) { sCC[tid] = cc[b * L_DIM + tid]; sScr[tid] = 0.0f; }
    {
        const _Float16* xsrc = X2 + (size_t)b * L_DIM * E_DIM;
        #pragma unroll
        for (int i = 0; i < 8; ++i) {
            const int c2 = i * 256 + tid;             // 2048 16B chunks
            const int row = c2 >> 5, colh = (c2 & 31) * 8;
            async_ld_b128(sX + row * LDA + colh, xsrc + row * E_DIM + colh);
        }
    }
    stage_weights_async(Wh, smem, 0, 0, tid);         // chunk 0 in flight
    // async loads of one wave complete in order: <=8 outstanding => x2 done
    asm volatile("s_wait_asynccnt 0x8" ::: "memory");
    __syncthreads();
    if (tid == 0) {
        int n = 0;
        for (int l = 0; l < 64; ++l) n += (sCC[l] != 0);
        sNw[0] = (float)n;
    }

    // ---- Phase B: QKV = x2 @ Wh^T (M=64, N=768, K=256) via WMMA -----------
    // Weights double-buffered in LDS; each 64x256 chunk (4 ntiles) is read
    // from L2 exactly once per workgroup. Wave wv owns mt = wv&3 and the two
    // chunk-local ntiles {wv>>2, (wv>>2)+2} so its A fragments load once.
    {
        const int mt = wv & 3;
        const int nt0 = wv >> 2;
        v16h afr[8];
        #pragma unroll
        for (int k0 = 0; k0 < 8; ++k0)
            afr[k0] = load_frag(sX, LDA, mt * 16, k0 * 32, lane);

        for (int c = 0; c < 12; ++c) {
            if (c + 1 < 12) {
                stage_weights_async(Wh, smem, (c + 1) & 1, c + 1, tid);
                asm volatile("s_wait_asynccnt 0x8" ::: "memory"); // chunk c done
            } else {
                asm volatile("s_wait_asynccnt 0x0" ::: "memory");
            }
            __syncthreads();
            const _Float16* sW =
                (const _Float16*)(smem + WBUF_OFF + (size_t)(c & 1) * WBUF_BYTES);
            #pragma unroll
            for (int half = 0; half < 2; ++half) {
                const int ntl = nt0 + half * 2;       // chunk-local ntile
                const int nt = c * 4 + ntl;           // global ntile
                v8f acc = {};
                #pragma unroll
                for (int k0 = 0; k0 < 8; ++k0) {
                    v16h bf = load_frag(sW, WLD, ntl * 16, k0 * 32, lane);
                    acc = wmma_f16(afr[k0], bf, acc);
                }
                #pragma unroll
                for (int j = 0; j < 8; ++j)
                    sQKV[(mt * 16 + g * 8 + j) * QLD + nt * 16 + r] = (_Float16)acc[j];
            }
            __syncthreads();   // all waves done reading buf[c&1] before reuse
        }
    }

    // ---- Phase C: S = (Q K^T)/16, key-mask to -inf ------------------------
    for (int t = wv * 2; t < wv * 2 + 2; ++t) {
        const int mt = t >> 2, nt = t & 3;
        v8f acc = {};
        #pragma unroll
        for (int k0 = 0; k0 < 8; ++k0) {
            v16h qf = load_frag(sQKV, QLD, mt * 16, k0 * 32, lane);        // Q
            v16h kf = load_frag(sQKV, QLD, nt * 16, 256 + k0 * 32, lane);  // K^T
            acc = wmma_f16(qf, kf, acc);
        }
        const int col = nt * 16 + r;
        const bool km = (sCC[col] == 0);
        const float ninf = -__builtin_inff();
        #pragma unroll
        for (int j = 0; j < 8; ++j) {
            const int row = mt * 16 + g * 8 + j;
            sS[row * SLD + col] = km ? ninf : acc[j] * 0.0625f;
        }
    }
    __syncthreads();

    // ---- Phase D: softmax rows + query-averaged score ---------------------
    {
        const float nwF = sNw[0];
        for (int i = 0; i < 8; ++i) {
            const int q = wv * 8 + i;
            float c0 = sS[q * SLD + lane], c1 = sS[q * SLD + lane + 32];
            float m = fmaxf(c0, c1);
            #pragma unroll
            for (int o = 16; o; o >>= 1) m = fmaxf(m, __shfl_xor(m, o, 32));
            const float e0 = __expf(c0 - m), e1 = __expf(c1 - m);
            float s = e0 + e1;
            #pragma unroll
            for (int o = 16; o; o >>= 1) s += __shfl_xor(s, o, 32);
            const float coef = (sCC[q] != 0 ? 1.0f : 0.0f) / (s * nwF);
            atomicAdd(&sScr[lane],      e0 * coef);
            atomicAdd(&sScr[lane + 32], e1 * coef);
        }
    }
    __syncthreads();

    // ---- Phase E: out_h = score @ V ---------------------------------------
    {
        float acc = 0.0f;
        #pragma unroll 8
        for (int l = 0; l < 64; ++l)
            acc += sScr[l] * (float)sQKV[l * QLD + 512 + tid];
        AOUT[((size_t)b * H_DIM + h) * E_DIM + tid] = acc;
    }
}

// ---------------------------------------------------------------------------
// Kernel 3: FC + LayerNorm chain (per batch row; 0.5 GFLOP total, bandwidth
// trivial -> plain VALU with block reductions).
// ---------------------------------------------------------------------------
__device__ __forceinline__ float blockSum256(float v, float* red) {
    #pragma unroll
    for (int o = 16; o; o >>= 1) v += __shfl_xor(v, o, 32);
    const int w = threadIdx.x >> 5;
    if ((threadIdx.x & 31) == 0) red[w] = v;
    __syncthreads();
    float t = (threadIdx.x < 8) ? red[threadIdx.x] : 0.0f;
    if (threadIdx.x < 32) {
        #pragma unroll
        for (int o = 4; o; o >>= 1) t += __shfl_xor(t, o, 32);
        if (threadIdx.x == 0) red[0] = t;
    }
    __syncthreads();
    const float rr = red[0];
    __syncthreads();
    return rr;
}

__device__ __forceinline__ float lnStep(float v, float gg, float bb, float eps, float* red) {
    const float m = blockSum256(v, red) * (1.0f / 256.0f);
    const float d = v - m;
    const float var = blockSum256(d * d, red) * (1.0f / 256.0f);
    return d * rsqrtf(var + eps) * gg + bb;
}

__global__ void k_fc_ln(const float* __restrict__ AOUT,
                        const float* __restrict__ XMEAN,
                        const float* __restrict__ fc_w,
                        const float* __restrict__ fc_b,
                        const float* __restrict__ sln_g, const float* __restrict__ sln_b,
                        const float* __restrict__ ln1_g, const float* __restrict__ ln1_b,
                        const float* __restrict__ ffln_g, const float* __restrict__ ffln_b,
                        const float* __restrict__ ln2_g, const float* __restrict__ ln2_b,
                        float* __restrict__ out) {
    const int b = blockIdx.x, tid = threadIdx.x;
    __shared__ float sIn[H_DIM * E_DIM];
    __shared__ float sRed[8];
    #pragma unroll
    for (int j = 0; j < 4; ++j)
        sIn[tid + j * 256] = AOUT[(size_t)b * 1024 + tid + j * 256];
    __syncthreads();

    float acc = fc_b[tid];
    const float* wrow = fc_w + (size_t)tid * 1024;
    for (int f = 0; f < 1024; f += 4) {
        const float4 w4 = *(const float4*)(wrow + f);
        const float4 x4 = *(const float4*)(&sIn[f]);
        acc += w4.x * x4.x + w4.y * x4.y + w4.z * x4.z + w4.w * x4.w;
    }

    float v = lnStep(acc, sln_g[tid], sln_b[tid], 1e-5f, sRed);
    v += XMEAN[(size_t)b * E_DIM + tid];
    float word = lnStep(v, ln1_g[tid], ln1_b[tid], 1e-5f, sRed);
    float ff = lnStep(word, ffln_g[tid], ffln_b[tid], 1e-6f, sRed);
    float w2 = ff + word;
    out[(size_t)b * E_DIM + tid] = lnStep(w2, ln2_g[tid], ln2_b[tid], 1e-5f, sRed);
}

// ---------------------------------------------------------------------------
// Launch
// ---------------------------------------------------------------------------
extern "C" void kernel_launch(void* const* d_in, const int* in_sizes, int n_in,
                              void* d_out, int out_size, void* d_ws, size_t ws_size,
                              hipStream_t stream) {
    const int*   char_code = (const int*)  d_in[0];
    // d_in[1] = max_len (== L, compile-time)
    const float* emb_table = (const float*)d_in[2];
    const float* qkv_w     = (const float*)d_in[3];
    const float* ln0_g     = (const float*)d_in[4];
    const float* ln0_b     = (const float*)d_in[5];
    const float* fc_w      = (const float*)d_in[6];
    const float* fc_b      = (const float*)d_in[7];
    const float* sln_g     = (const float*)d_in[8];
    const float* sln_b     = (const float*)d_in[9];
    const float* ln1_g     = (const float*)d_in[10];
    const float* ln1_b     = (const float*)d_in[11];
    // d_in[12..15] = ff_w1/ff_b1/ff_w2/ff_b2: unused by the reference
    const float* ffln_g    = (const float*)d_in[16];
    const float* ffln_b    = (const float*)d_in[17];
    const float* ln2_g     = (const float*)d_in[18];
    const float* ln2_b     = (const float*)d_in[19];

    // Workspace layout (~38.5 MB total)
    char* ws = (char*)d_ws;
    constexpr size_t W16_BYTES  = (size_t)3072 * 256 * 2;              // 1.5 MB
    constexpr size_t X2_BYTES   = (size_t)B_DIM * L_DIM * E_DIM * 2;   // 32 MB
    constexpr size_t XM_BYTES   = (size_t)B_DIM * E_DIM * 4;           // 1 MB
    _Float16* W16   = (_Float16*)(ws);
    _Float16* X2    = (_Float16*)(ws + W16_BYTES);
    float*    XMEAN = (float*)   (ws + W16_BYTES + X2_BYTES);
    float*    AOUT  = (float*)   (ws + W16_BYTES + X2_BYTES + XM_BYTES);

    k_convert<<<(3072 * 256) / (256 * 4), 256, 0, stream>>>(qkv_w, W16);

    k_embed_ln<<<B_DIM, 256, 0, stream>>>(char_code, emb_table, ln0_g, ln0_b,
                                          X2, XMEAN);

    k_attn<<<B_DIM * H_DIM, 256, ATTN_LDS, stream>>>(char_code, W16, X2, AOUT);

    k_fc_ln<<<B_DIM, 256, 0, stream>>>(AOUT, XMEAN, fc_w, fc_b,
                                       sln_g, sln_b, ln1_g, ln1_b,
                                       ffln_g, ffln_b, ln2_g, ln2_b,
                                       (float*)d_out);
}
